// ImageRearrangement_43379169689665
// MI455X (gfx1250) — matrix operand order repaired and checked by reading, not compiled, not run on hardware
//
#include <hip/hip_runtime.h>
#include <cstdint>
#include <cstddef>

// ---------------- problem constants ----------------
constexpr int Bc   = 2;
constexpr int Sc   = 1024;
constexpr int HIDc = 4096;
constexpr int NHc  = 32;
constexpr int HDc  = 128;
constexpr int EDGEc = 24;
constexpr int KWc  = 12;
constexpr int PEc  = 13;     // 1 + 24 - 12
constexpr int VTc  = 576;    // 24*24
constexpr int OFFc = 5;
constexpr int UPNc = 512;    // HD*SR*SR

// ---------------- CDNA5 WMMA types ----------------
typedef __attribute__((ext_vector_type(16))) __bf16 v16bf;
typedef __attribute__((ext_vector_type(8)))  float  v8f;

__device__ __forceinline__ uint16_t f2bf(float x) {
    uint32_t u = __float_as_uint(x);
    uint32_t r = (u + 0x7FFFu + ((u >> 16) & 1u)) >> 16;   // round-to-nearest-even
    return (uint16_t)r;
}

// A fragment (16x32 bf16, M x K). rowptr = start of this lane's 32-elem K chunk
// for row (lane&15). ISA layout: lane<16 holds K{0..7,16..23}, lane>=16 holds
// K{8..15,24..31}.
__device__ __forceinline__ v16bf load_afrag(const uint16_t* rowptr, int lane_hi) {
    union { v16bf v; uint4 q[2]; } f;
    f.q[0] = *(const uint4*)(rowptr + 8 * lane_hi);
    f.q[1] = *(const uint4*)(rowptr + 16 + 8 * lane_hi);
    return f.v;
}

// B fragment (32x16 bf16, K x N) sourced from N-major rows of length K
// (i.e. B[k][n] = W[n][k]). lane<16 holds K{0..7,16..23} of column n=lane, lane>=16
// holds K{8..15,24..31} of column n=lane-16.
__device__ __forceinline__ v16bf load_bfrag(const uint16_t* rowptr, int lane_hi) {
    union { v16bf v; uint4 q[2]; } f;
    f.q[0] = *(const uint4*)(rowptr + 8 * lane_hi);
    f.q[1] = *(const uint4*)(rowptr + 16 + 8 * lane_hi);
    return f.v;
}

// CDNA5 async global->LDS copy, 16 bytes per lane; tracked by ASYNCcnt.
// (generic pointers to __shared__ carry the LDS byte offset in their low 32 bits)
__device__ __forceinline__ void async_load_b128(const void* gaddr, void* lds) {
    uint32_t l = (uint32_t)(uintptr_t)lds;
    asm volatile("global_load_async_to_lds_b128 %0, %1, off"
                 :: "v"(l), "v"(gaddr) : "memory");
}
#define S_WAIT_ASYNCCNT(n) asm volatile("s_wait_asynccnt %0" :: "i"(n) : "memory")

// ---------------- conversion / init kernels ----------------
__global__ void cvt_dual_kernel(const float* __restrict__ src,
                                uint16_t* __restrict__ d0,
                                uint16_t* __restrict__ d1, size_t n) {
    size_t i = (size_t)blockIdx.x * blockDim.x + threadIdx.x;
    if (i < n) { uint16_t v = f2bf(src[i]); d0[i] = v; d1[i] = v; }
}

__global__ void cvt_kernel(const float* __restrict__ src,
                           uint16_t* __restrict__ dst, size_t n) {
    size_t i = (size_t)blockIdx.x * blockDim.x + threadIdx.x;
    if (i < n) dst[i] = f2bf(src[i]);
}

__global__ void fill_kernel(float* __restrict__ p, float v, size_t n) {
    size_t i = (size_t)blockIdx.x * blockDim.x + threadIdx.x;
    if (i < n) p[i] = v;
}

// ---------------- attention: sum of softmax rows over unmasked q ----------------
// grid (28, NH, B), block = 32 (one wave). Each block: 16 unmasked query rows,
// full 1024 keys via WMMA, then softmax stats + atomic accumulate into
// image_attn (pre-initialized to 576/1024 for the uniform masked rows).
__global__ void attn_kernel(const uint16_t* __restrict__ hsb,
                            float* __restrict__ image_attn) {
    __shared__ float sc[16][1024];          // 64 KB
    const int t = blockIdx.x, h = blockIdx.y, b = blockIdx.z;
    const int lane = threadIdx.x;
    const int r = lane & 15, hh = lane >> 4;

    // A fragments: 4 K-chunks of this lane's query row, kept in registers
    int u = t * 16 + r;                     // unmasked row index 0..447
    int q = (u < OFFc) ? u : u + VTc;
    const uint16_t* ap = hsb + ((size_t)b * Sc + q) * HIDc + h * HDc;
    v16bf afrag[4];
#pragma unroll
    for (int c = 0; c < 4; ++c) afrag[c] = load_afrag(ap + 32 * c, hh);

    const float scale = 0.08838834764831845f;   // 1/sqrt(128)
    for (int kt = 0; kt < 64; ++kt) {
        const uint16_t* kb = hsb + ((size_t)b * Sc + kt * 16 + r) * HIDc + h * HDc;
        v8f acc = {};
#pragma unroll
        for (int c = 0; c < 4; ++c) {
            v16bf bfrag = load_bfrag(kb + 32 * c, hh);
            acc = __builtin_amdgcn_wmma_f32_16x16x32_bf16(
                false, afrag[c], false, bfrag, (short)0, acc, false, false);
        }
        int col = kt * 16 + r;
#pragma unroll
        for (int j = 0; j < 8; ++j) sc[j + 8 * hh][col] = acc[j] * scale;
    }

    // per-row softmax statistics (wave-level)
    float rmax[16], rinv[16];
#pragma unroll 1
    for (int m = 0; m < 16; ++m) {
        float mx = -3.4e38f;
        for (int kk = lane; kk < 1024; kk += 32) mx = fmaxf(mx, sc[m][kk]);
        for (int o = 16; o > 0; o >>= 1) mx = fmaxf(mx, __shfl_xor(mx, o, 32));
        float sm = 0.f;
        for (int kk = lane; kk < 1024; kk += 32) sm += __expf(sc[m][kk] - mx);
        for (int o = 16; o > 0; o >>= 1) sm += __shfl_xor(sm, o, 32);
        rmax[m] = mx; rinv[m] = 1.0f / sm;
    }

    // accumulate softmax mass landing on image keys 5..580
    float* ia = image_attn + ((size_t)b * NHc + h) * VTc;
    for (int kk = lane; kk < VTc; kk += 32) {
        float a = 0.f;
#pragma unroll
        for (int m = 0; m < 16; ++m) a += __expf(sc[m][OFFc + kk] - rmax[m]) * rinv[m];
        atomicAdd(ia + kk, a);
    }
}

// ---------------- per-head window selection ----------------
// grid (NH, B), block 256.
__global__ void select_kernel(const float* __restrict__ image_attn,
                              int* __restrict__ selr, int* __restrict__ selc,
                              float* __restrict__ selw) {
    __shared__ float s_ia[VTc];
    __shared__ float s_soft[VTc];
    __shared__ float rbuf[256];
    __shared__ int   ibuf[256];
    const int tid = threadIdx.x;
    const int head = blockIdx.y * NHc + blockIdx.x;
    const float* ia = image_attn + (size_t)head * VTc;

    for (int i = tid; i < VTc; i += 256) s_ia[i] = ia[i];
    __syncthreads();

    // softmax over 576
    float mx = -3.4e38f;
    for (int i = tid; i < VTc; i += 256) mx = fmaxf(mx, s_ia[i]);
    rbuf[tid] = mx; __syncthreads();
    for (int s = 128; s > 0; s >>= 1) {
        if (tid < s) rbuf[tid] = fmaxf(rbuf[tid], rbuf[tid + s]);
        __syncthreads();
    }
    mx = rbuf[0]; __syncthreads();
    float sm = 0.f;
    for (int i = tid; i < VTc; i += 256) sm += __expf(s_ia[i] - mx);
    rbuf[tid] = sm; __syncthreads();
    for (int s = 128; s > 0; s >>= 1) {
        if (tid < s) rbuf[tid] += rbuf[tid + s];
        __syncthreads();
    }
    float inv = 1.0f / rbuf[0]; __syncthreads();
    for (int i = tid; i < VTc; i += 256) s_soft[i] = __expf(s_ia[i] - mx) * inv;
    __syncthreads();

    // 12x12 valid avg-pool argmax on RAW image_attn (div by 144 irrelevant)
    float best = -3.4e38f; int bidx = 1 << 30;
    if (tid < PEc * PEc) {
        int r = tid / PEc, c = tid % PEc;
        float s = 0.f;
        for (int i = 0; i < KWc; ++i)
            for (int j = 0; j < KWc; ++j) s += s_ia[(r + i) * EDGEc + (c + j)];
        best = s; bidx = tid;
    }
    rbuf[tid] = best; ibuf[tid] = bidx; __syncthreads();
    for (int s = 128; s > 0; s >>= 1) {
        if (tid < s) {
            float o = rbuf[tid + s]; int oi = ibuf[tid + s];
            if (o > rbuf[tid] || (o == rbuf[tid] && oi < ibuf[tid])) {
                rbuf[tid] = o; ibuf[tid] = oi;
            }
        }
        __syncthreads();
    }
    if (tid == 0) {
        int idx = ibuf[0];
        int r = idx / PEc, c = idx % PEc;
        float w = 0.f;
        for (int i = 0; i < KWc; ++i)
            for (int j = 0; j < KWc; ++j) w += s_soft[(r + i) * EDGEc + (c + j)];
        selr[head] = r; selc[head] = c; selw[head] = w;
    }
}

// ---------------- window up-projection GEMM (144 x 512 x 128 per head) ---------
// grid (72, NH, B), block 128 (4 waves). Each wave: one 16x16 output tile.
// Fused: scale by w, pixel-shuffle 12x12x(2,2,128) -> 24x24x128, bf16 scatter
// into the image rows of hs_in.
__global__ void win_gemm_kernel(const uint16_t* __restrict__ hsb,
                                const uint16_t* __restrict__ wupb,
                                const int* __restrict__ selr,
                                const int* __restrict__ selc,
                                const float* __restrict__ selw,
                                uint16_t* __restrict__ hsin) {
    const int h = blockIdx.y, b = blockIdx.z;
    const int head = b * NHc + h;
    const int mt = blockIdx.x >> 3;            // 0..8  (9 m-tiles of 16, M=144)
    const int ng = blockIdx.x & 7;             // 0..7
    const int wave = threadIdx.x >> 5;         // 0..3
    const int lane = threadIdx.x & 31;
    const int nt = ng * 4 + wave;              // 0..31 (N=512)
    const int r = lane & 15, hh = lane >> 4;

    const int r0 = selr[head], c0 = selc[head];
    const float w = selw[head];

    const int m = mt * 16 + r;                 // 0..143
    const int wi = m / KWc, wj = m % KWc;
    const int seq = OFFc + (r0 + wi) * EDGEc + (c0 + wj);
    const uint16_t* ap = hsb + ((size_t)b * Sc + seq) * HIDc + h * HDc;
    const uint16_t* bp = wupb + (size_t)(nt * 16 + r) * HDc;

    v8f acc = {};
#pragma unroll
    for (int c = 0; c < 4; ++c) {
        v16bf a = load_afrag(ap + 32 * c, hh);
        v16bf bf = load_bfrag(bp + 32 * c, hh);
        acc = __builtin_amdgcn_wmma_f32_16x16x32_bf16(
            false, a, false, bf, (short)0, acc, false, false);
    }

    const int e = nt * 16 + r;                 // output channel 0..511
    const int ua = e >> 8, ub = (e >> 7) & 1, d = e & 127;
#pragma unroll
    for (int jj = 0; jj < 8; ++jj) {
        int mrow = mt * 16 + jj + 8 * hh;      // 0..143
        int ii = mrow / KWc, jc = mrow % KWc;
        int v = ii * 48 + ua * 24 + jc * 2 + ub;   // pixel-shuffle position
        hsin[((size_t)b * Sc + OFFc + v) * HIDc + h * HDc + d] = f2bf(acc[jj] * w);
    }
}

// ---------------- final GEMM: out = residual + silu(hs_in @ W_in^T) -----------
// M=2048, N=4096, K=4096. Block 256 thr (8 waves), tile 64(M) x 128(N).
// K is consumed in 64-wide slices, double-buffered in LDS and staged with
// CDNA5 async global->LDS b128 copies (ASYNCcnt) so the copy of slice k+1
// overlaps the 8 WMMAs/wave of slice k. 48 KB LDS (of 320 KB/WGP).
__global__ void final_gemm_kernel(const uint16_t* __restrict__ hsin,
                                  const uint16_t* __restrict__ winb,
                                  const float* __restrict__ residual,
                                  float* __restrict__ out) {
    __shared__ uint16_t As[2][64][64];     // 16 KB
    __shared__ uint16_t Bs[2][128][64];    // 32 KB
    const int m0 = blockIdx.x * 64;
    const int n0 = blockIdx.y * 128;
    const int tid = threadIdx.x;
    const int wave = tid >> 5, lane = tid & 31;
    const int mw = wave >> 2, nw = wave & 3;
    const int hh = lane >> 4;

    // 6 async b128 copies per thread per 64-wide K slice
    auto issue_slice = [&](int k0, int bufi) {
#pragma unroll
        for (int rr = 0; rr < 2; ++rr) {              // A: 64x64 = 512 chunks
            int idx = tid + rr * 256;
            int row = idx >> 3, cg = idx & 7;
            async_load_b128(&hsin[(size_t)(m0 + row) * HIDc + k0 + cg * 8],
                            &As[bufi][row][cg * 8]);
        }
#pragma unroll
        for (int rr = 0; rr < 4; ++rr) {              // B: 128x64 = 1024 chunks
            int idx = tid + rr * 256;
            int row = idx >> 3, cg = idx & 7;
            async_load_b128(&winb[(size_t)(n0 + row) * HIDc + k0 + cg * 8],
                            &Bs[bufi][row][cg * 8]);
        }
    };

    v8f acc[2][2] = {};
    issue_slice(0, 0);
    for (int ks = 0; ks < 64; ++ks) {
        const int cur = ks & 1;
        if (ks + 1 < 64) {
            issue_slice((ks + 1) * 64, cur ^ 1);
            S_WAIT_ASYNCCNT(6);    // this wave's slice-ks copies retired (in order)
        } else {
            S_WAIT_ASYNCCNT(0);
        }
        __syncthreads();           // all waves' copies of slice ks are in LDS

#pragma unroll
        for (int kc = 0; kc < 2; ++kc) {
            v16bf a[2], bf[2];
#pragma unroll
            for (int t = 0; t < 2; ++t) {
                a[t]  = load_afrag(&As[cur][mw * 32 + t * 16 + (lane & 15)][kc * 32], hh);
                bf[t] = load_bfrag(&Bs[cur][nw * 32 + t * 16 + (lane & 15)][kc * 32], hh);
            }
#pragma unroll
            for (int wm = 0; wm < 2; ++wm)
#pragma unroll
                for (int wn = 0; wn < 2; ++wn)
                    acc[wm][wn] = __builtin_amdgcn_wmma_f32_16x16x32_bf16(
                        false, a[wm], false, bf[wn], (short)0, acc[wm][wn], false, false);
        }
        __syncthreads();           // safe to overwrite this buffer next round
    }

#pragma unroll
    for (int wm = 0; wm < 2; ++wm)
#pragma unroll
        for (int wn = 0; wn < 2; ++wn) {
            int gn = n0 + nw * 32 + wn * 16 + (lane & 15);
#pragma unroll
            for (int jj = 0; jj < 8; ++jj) {
                int gm = m0 + mw * 32 + wm * 16 + jj + 8 * hh;
                float x = acc[wm][wn][jj];
                float s = x / (1.0f + __expf(-x));      // silu
                out[(size_t)gm * HIDc + gn] =
                    residual[(size_t)gm * HIDc + gn] + s;
            }
        }
}

// ---------------- host launcher ----------------
extern "C" void kernel_launch(void* const* d_in, const int* in_sizes, int n_in,
                              void* d_out, int out_size, void* d_ws, size_t ws_size,
                              hipStream_t stream) {
    (void)in_sizes; (void)n_in; (void)out_size; (void)ws_size;
    const float* hidden = (const float*)d_in[0];
    const float* W_in   = (const float*)d_in[1];
    const float* W_up   = (const float*)d_in[2];
    float* out = (float*)d_out;

    char* ws = (char*)d_ws;
    size_t off = 0;
    auto take = [&](size_t bytes) -> char* {
        char* p = ws + off;
        off += (bytes + 255) & ~(size_t)255;
        return p;
    };

    const size_t NHS = (size_t)Bc * Sc * HIDc;                 // 8,388,608
    uint16_t* hsb  = (uint16_t*)take(NHS * 2);                 // hidden bf16
    uint16_t* hsin = (uint16_t*)take(NHS * 2);                 // GEMM input bf16
    uint16_t* winb = (uint16_t*)take((size_t)HIDc * HIDc * 2); // W_in bf16
    uint16_t* wupb = (uint16_t*)take((size_t)UPNc * HDc * 2);  // W_up bf16
    float* image_attn = (float*)take((size_t)Bc * NHc * VTc * 4);
    int*   selr = (int*)take(Bc * NHc * 4);
    int*   selc = (int*)take(Bc * NHc * 4);
    float* selw = (float*)take(Bc * NHc * 4);

    cvt_dual_kernel<<<dim3((unsigned)(NHS / 256)), 256, 0, stream>>>(hidden, hsb, hsin, NHS);
    cvt_kernel<<<dim3((unsigned)(((size_t)HIDc * HIDc) / 256)), 256, 0, stream>>>(
        W_in, winb, (size_t)HIDc * HIDc);
    cvt_kernel<<<dim3((UPNc * HDc) / 256), 256, 0, stream>>>(W_up, wupb, (size_t)UPNc * HDc);
    // masked query rows (576 of them) each contribute uniform 1/1024 to every key
    fill_kernel<<<dim3((Bc * NHc * VTc + 255) / 256), 256, 0, stream>>>(
        image_attn, (float)VTc / (float)Sc, (size_t)Bc * NHc * VTc);

    attn_kernel<<<dim3(28, NHc, Bc), 32, 0, stream>>>(hsb, image_attn);
    select_kernel<<<dim3(NHc, Bc), 256, 0, stream>>>(image_attn, selr, selc, selw);
    win_gemm_kernel<<<dim3(72, NHc, Bc), 128, 0, stream>>>(
        hsb, wupb, selr, selc, selw, hsin);
    final_gemm_kernel<<<dim3((Bc * Sc) / 64, HIDc / 128), 256, 0, stream>>>(
        hsin, winb, hidden, out);
}
